// GRU_3229815407360
// MI455X (gfx1250) — compile-verified, matrix operand order checked by reading, not run on hardware
//
#include <hip/hip_runtime.h>

typedef unsigned short u16;
typedef unsigned int   u32;

typedef __bf16 bf16x16 __attribute__((ext_vector_type(16)));
typedef float  f32x8   __attribute__((ext_vector_type(8)));
typedef int    i32x4   __attribute__((ext_vector_type(4)));

union FragBF {
    uint4   u[2];   // 32 bytes = 16 bf16
    bf16x16 v;
};

#define T_LEN   8192
#define HDIM    1024
#define PLANE   (T_LEN * HDIM)          // elements per projection plane
#define NWG2    64                      // workgroups in scan kernel
#define RPW     16                      // rows of h per workgroup (1024/64)

// ------------------------------------------------ async global->LDS helpers
#if defined(__has_builtin)
#if __has_builtin(__builtin_amdgcn_global_load_async_to_lds_b128)
#define HAVE_ASYNC_LDS 1
#endif
#endif
#ifndef HAVE_ASYNC_LDS
#define HAVE_ASYNC_LDS 0
#endif

__device__ __forceinline__ void async_g2l_b128(void* ldst, const void* gsrc) {
#if HAVE_ASYNC_LDS
    __builtin_amdgcn_global_load_async_to_lds_b128(
        (__attribute__((address_space(1))) i32x4*)gsrc,
        (__attribute__((address_space(3))) i32x4*)ldst,
        0, 0);
#else
    *(uint4*)ldst = *(const uint4*)gsrc;
#endif
}

__device__ __forceinline__ void wait_async_all() {
#if HAVE_ASYNC_LDS
#if __has_builtin(__builtin_amdgcn_s_wait_asynccnt)
    __builtin_amdgcn_s_wait_asynccnt(0);
#endif
#endif
}

// ---------------------------------------------------------------- helpers
__device__ __forceinline__ u16 f2bf(float f) {
    u32 u = __float_as_uint(f);
    return (u16)((u + 0x7FFFu + ((u >> 16) & 1u)) >> 16);   // RNE
}

__device__ __forceinline__ float2 bfpair(u32 w) {
    float2 r;
    r.x = __uint_as_float(w << 16);
    r.y = __uint_as_float(w & 0xFFFF0000u);
    return r;
}

__device__ __forceinline__ float sigmoidf(float x) {
    return 1.0f / (1.0f + __expf(-x));
}

// ---------------------------------------------------------------- convert
__global__ void f32_to_bf16_kernel(const float* __restrict__ src,
                                   u16* __restrict__ dst, int n) {
    int i = blockIdx.x * blockDim.x + threadIdx.x;
    if (i < n) dst[i] = f2bf(src[i]);
}

// ---------------------------------------------------------------- phase 1
// P_m[t, h] = sum_e xb[t, e] * Ub_m[h, e] + bias_m[h]
// Block tile: 256 (M) x 64 (N); 8 waves, each wave owns a 32-row M slice
// (2 A-frags) x 4 WMMA column tiles. The shared 64x32 B-tile is staged into
// a double-buffered LDS tile with async global->LDS copies so the next
// tile's DMA overlaps the current tile's WMMAs.
__global__ __launch_bounds__(256)
void proj_gemm(const u16* __restrict__ xb, const u16* __restrict__ Ub,
               const float* __restrict__ Bu, const float* __restrict__ Br,
               const float* __restrict__ Bc, float* __restrict__ P) {
    __shared__ u16 btile[2][64 * 32];   // 2 x 4 KB, layout [n][k]

    const int mat = blockIdx.z;
    const u16* __restrict__ Um = Ub + (size_t)mat * (HDIM * HDIM);
    const float* __restrict__ bias = (mat == 0) ? Bu : ((mat == 1) ? Br : Bc);
    float* __restrict__ Pm = P + (size_t)mat * PLANE;

    const int tid   = threadIdx.x;
    const int wave  = tid >> 5;
    const int lane  = tid & 31;
    const int lrow  = lane & 15;
    const int khalf = (lane >> 4) * 8;          // 0 or 8 (ISA A/B frag layout)

    const int mBase = blockIdx.x * 256 + wave * 32;
    const int nBase = blockIdx.y * 64;

    // per-thread staging slot: 16 bytes of the 4 KB B tile
    const int sn = tid >> 2;                    // 0..63  (column)
    const int sk = (tid & 3) * 8;               // 0,8,16,24 (k chunk)
    const u16* sgsrc = Um + (size_t)(nBase + sn) * HDIM + sk;
    u16* sldst[2] = { &btile[0][sn * 32 + sk], &btile[1][sn * 32 + sk] };

    const u16* aptr0 = xb + (size_t)(mBase + lrow) * HDIM + khalf;
    const u16* aptr1 = xb + (size_t)(mBase + 16 + lrow) * HDIM + khalf;

    f32x8 acc[2][4];
#pragma unroll
    for (int i = 0; i < 2; ++i)
#pragma unroll
        for (int j = 0; j < 4; ++j)
            acc[i][j] = (f32x8){0.f,0.f,0.f,0.f,0.f,0.f,0.f,0.f};

    // prologue: stage k=0 tile
    async_g2l_b128(sldst[0], sgsrc);
    wait_async_all();
    __syncthreads();

    int cur = 0;
    for (int kt = 0; kt < 32; ++kt) {
        const int k = kt * 32;
        if (kt != 31)
            async_g2l_b128(sldst[cur ^ 1], sgsrc + k + 32);   // prefetch next

        FragBF a0, a1;
        a0.u[0] = *(const uint4*)(aptr0 + k);
        a0.u[1] = *(const uint4*)(aptr0 + k + 16);
        a1.u[0] = *(const uint4*)(aptr1 + k);
        a1.u[1] = *(const uint4*)(aptr1 + k + 16);

        const u16* bt = btile[cur];
#pragma unroll
        for (int j = 0; j < 4; ++j) {
            FragBF b;
            b.u[0] = *(const uint4*)(bt + (j * 16 + lrow) * 32 + khalf);
            b.u[1] = *(const uint4*)(bt + (j * 16 + lrow) * 32 + khalf + 16);
            acc[0][j] = __builtin_amdgcn_wmma_f32_16x16x32_bf16(
                            false, a0.v, false, b.v, (short)0, acc[0][j], false, false);
            acc[1][j] = __builtin_amdgcn_wmma_f32_16x16x32_bf16(
                            false, a1.v, false, b.v, (short)0, acc[1][j], false, false);
        }

        wait_async_all();     // next tile landed (our wave's copies)
        __syncthreads();      // everyone done reading cur + all copies visible
        cur ^= 1;
    }

    // C/D layout: lanes 0-15 -> M = v, lanes 16-31 -> M = 8 + v ; N = lane&15
    const int ncol = lane & 15;
    const int moff = (lane >> 4) * 8;
#pragma unroll
    for (int j = 0; j < 4; ++j) {
        const int n  = nBase + j * 16 + ncol;
        const float bv = bias[n];
#pragma unroll
        for (int v = 0; v < 8; ++v) {
            Pm[(size_t)(mBase + moff + v) * HDIM + n]      = acc[0][j][v] + bv;
            Pm[(size_t)(mBase + 16 + moff + v) * HDIM + n] = acc[1][j][v] + bv;
        }
    }
}

// ---------------------------------------------------------------- grid sync
__device__ __forceinline__ void grid_sync(int* cnt, int* gen, int nwg, int step) {
    __syncthreads();
    if (threadIdx.x == 0) {
        __threadfence();
        int prev = __hip_atomic_fetch_add(cnt, 1, __ATOMIC_ACQ_REL,
                                          __HIP_MEMORY_SCOPE_AGENT);
        if (prev == nwg - 1) {
            __hip_atomic_store(cnt, 0, __ATOMIC_RELAXED, __HIP_MEMORY_SCOPE_AGENT);
            __hip_atomic_fetch_add(gen, 1, __ATOMIC_RELEASE,
                                   __HIP_MEMORY_SCOPE_AGENT);
        } else {
            while (__hip_atomic_load(gen, __ATOMIC_ACQUIRE,
                                     __HIP_MEMORY_SCOPE_AGENT) <= step)
                __builtin_amdgcn_s_sleep(1);
        }
    }
    __syncthreads();
}

// ---------------------------------------------------------------- phase 2
// Persistent scan. 64 WGs x 256 threads. Each WG keeps a 16-row slice of the
// three recurrent matrices (bf16) resident in LDS (96 KB) plus h (4 KB).
// h is double-buffered in global memory; one grid barrier per timestep.
__global__ __launch_bounds__(256)
void gru_scan(const float* __restrict__ P, const u16* __restrict__ Wb,
              float* __restrict__ hg, int* __restrict__ sync,
              float* __restrict__ out) {
    extern __shared__ unsigned char smem[];
    u16*   wl = (u16*)smem;                              // [3][16][1024] bf16
    float* hs = (float*)(smem + 3 * RPW * HDIM * 2);     // [1024]

    const int tid   = threadIdx.x;
    const int rbase = blockIdx.x * RPW;

    // Stage this WG's 16-row slice of Wu/Wr/W into LDS (6144 uint4s total)
    // via async global->LDS copies.
    for (int idx = tid; idx < 6144; idx += 256) {
        const int m   = idx >> 11;         // / 2048
        const int rem = idx & 2047;
        const int row = rem >> 7;          // / 128
        const int q   = rem & 127;
        const uint4* src =
            (const uint4*)(Wb + ((size_t)m << 20) + (size_t)(rbase + row) * HDIM) + q;
        async_g2l_b128(((uint4*)wl) + idx, src);
    }
    wait_async_all();
    for (int i = tid; i < HDIM; i += 256) hs[i] = 0.0f;   // h_0 = 0
    __syncthreads();

    const int row  = tid >> 4;            // 0..15 local row
    const int l16  = tid & 15;            // 16 lanes cooperate per row
    const int grow = rbase + row;
    const u32* w_u = (const u32*)(wl + 0 * RPW * HDIM + row * HDIM);
    const u32* w_r = (const u32*)(wl + 1 * RPW * HDIM + row * HDIM);
    const u32* w_c = (const u32*)(wl + 2 * RPW * HDIM + row * HDIM);
    const float2* h2 = (const float2*)hs;

    int* cnt = sync;
    int* gen = sync + 1;

    for (int t = 0; t < T_LEN; ++t) {
        float du = 0.f, dr = 0.f, dc = 0.f;
#pragma unroll 4
        for (int j = 0; j < 32; ++j) {
            const int p = l16 + (j << 4);          // stride-16 pair index
            const float2 h = h2[p];
            const float2 a = bfpair(w_u[p]);
            const float2 b = bfpair(w_r[p]);
            const float2 c = bfpair(w_c[p]);
            du = fmaf(a.x, h.x, fmaf(a.y, h.y, du));
            dr = fmaf(b.x, h.x, fmaf(b.y, h.y, dr));
            dc = fmaf(c.x, h.x, fmaf(c.y, h.y, dc));
        }
        // reduce across the 16 cooperating lanes (stays inside wave32 halves)
#pragma unroll
        for (int m = 8; m >= 1; m >>= 1) {
            du += __shfl_xor(du, m, 16);
            dr += __shfl_xor(dr, m, 16);
            dc += __shfl_xor(dc, m, 16);
        }
        if (l16 == 0) {
            const size_t tH = (size_t)t << 10;
            const float au  = P[tH + grow] + du;
            const float ar  = P[(size_t)PLANE + tH + grow] + dr;
            const float acd = P[(size_t)2 * PLANE + tH + grow];
            const float u    = sigmoidf(au);
            const float r    = sigmoidf(ar);
            const float cand = sigmoidf(acd + r * dc);
            const float hn   = u * hs[grow] + (1.0f - u) * cand;
            out[HDIM + tH + grow] = hn;                       // outputs[t]
            hg[(((t + 1) & 1) << 10) | grow] = hn;            // double buffer
            if (t == T_LEN - 1) out[grow] = hn;               // h_final
        }
        grid_sync(cnt, gen, (int)gridDim.x, t);
        // refresh full h vector from the buffer written this step
        ((float4*)hs)[tid] = ((const float4*)(hg + (((t + 1) & 1) << 10)))[tid];
        __syncthreads();
    }
}

// ---------------------------------------------------------------- launch
extern "C" void kernel_launch(void* const* d_in, const int* in_sizes, int n_in,
                              void* d_out, int out_size, void* d_ws, size_t ws_size,
                              hipStream_t stream) {
    (void)in_sizes; (void)n_in; (void)out_size; (void)ws_size;

    const float* x  = (const float*)d_in[0];
    const float* Uu = (const float*)d_in[1];
    const float* Wu = (const float*)d_in[2];
    const float* Bu = (const float*)d_in[3];
    const float* Ur = (const float*)d_in[4];
    const float* Wr = (const float*)d_in[5];
    const float* Br = (const float*)d_in[6];
    const float* U  = (const float*)d_in[7];
    const float* W  = (const float*)d_in[8];
    const float* B  = (const float*)d_in[9];
    float* out = (float*)d_out;

    unsigned char* ws = (unsigned char*)d_ws;
    size_t off = 0;
    u16*   xb = (u16*)(ws + off);  off += (size_t)T_LEN * HDIM * 2;       // 16 MB
    u16*   Ub = (u16*)(ws + off);  off += (size_t)3 * HDIM * HDIM * 2;    //  6 MB
    u16*   Wb = (u16*)(ws + off);  off += (size_t)3 * HDIM * HDIM * 2;    //  6 MB
    float* P  = (float*)(ws + off); off += (size_t)3 * PLANE * 4;         // 96 MB
    float* hg = (float*)(ws + off); off += (size_t)2 * HDIM * 4;          //  8 KB
    int* sync = (int*)(ws + off);   off += 256;

    // deterministic barrier state per launch (capturable)
    (void)hipMemsetAsync(sync, 0, 256, stream);

    const int NM = HDIM * HDIM;
    f32_to_bf16_kernel<<<(T_LEN * HDIM + 255) / 256, 256, 0, stream>>>(x, xb, T_LEN * HDIM);
    f32_to_bf16_kernel<<<(NM + 255) / 256, 256, 0, stream>>>(Uu, Ub + 0 * (size_t)NM, NM);
    f32_to_bf16_kernel<<<(NM + 255) / 256, 256, 0, stream>>>(Ur, Ub + 1 * (size_t)NM, NM);
    f32_to_bf16_kernel<<<(NM + 255) / 256, 256, 0, stream>>>(U,  Ub + 2 * (size_t)NM, NM);
    f32_to_bf16_kernel<<<(NM + 255) / 256, 256, 0, stream>>>(Wu, Wb + 0 * (size_t)NM, NM);
    f32_to_bf16_kernel<<<(NM + 255) / 256, 256, 0, stream>>>(Wr, Wb + 1 * (size_t)NM, NM);
    f32_to_bf16_kernel<<<(NM + 255) / 256, 256, 0, stream>>>(W,  Wb + 2 * (size_t)NM, NM);

    dim3 ggrid(T_LEN / 256, HDIM / 64, 3);
    proj_gemm<<<ggrid, 256, 0, stream>>>(xb, Ub, Bu, Br, B, P);

    const int smem_bytes = 3 * RPW * HDIM * 2 + HDIM * 4;   // 102,400 B
    (void)hipFuncSetAttribute((const void*)gru_scan,
                              hipFuncAttributeMaxDynamicSharedMemorySize, smem_bytes);
    gru_scan<<<NWG2, 256, smem_bytes, stream>>>(P, Wb, hg, sync, out);
}
